// PrototypicalLayer_3384434229451
// MI455X (gfx1250) — compile-verified, matrix operand order checked by reading, not run on hardware
//
#include <hip/hip_runtime.h>

// ---------------------------------------------------------------------------
// PrototypicalLayer: p = softmax_k( -||x_b - c_k||^2 )
//   = softmax_k( 2*(x.c^T)[b,k] - ||c_k||^2 )        (||x_b||^2 cancels)
//
// Pass 1 (tiny): clusters -> bf16 hi/lo planes in d_ws (L2-resident, 2 MB)
//                plus exact fp32 ||c_k||^2.
// Pass 2: fused GEMM via bf16 hi/lo split WMMA (3x v_wmma_f32_16x16x32_bf16,
//         ~fp32 accuracy) + in-workgroup row softmax.
//         Each WG: 512 threads = 16 waves, 32 x-rows x all 1024 clusters.
//         Each wave: 32x64 slab (2 M-tiles x 4 N-tiles) so every B fragment
//         feeds 6 WMMAs.
// ---------------------------------------------------------------------------

#define BDIM   65536
#define DDIM   512
#define NCL    1024
#define KC     32
#define NSTEPS (DDIM / KC)   // 16
#define MROWS  32            // x-rows per workgroup
#define NWAVES 16            // waves per workgroup (512 threads)

typedef __attribute__((ext_vector_type(16))) __bf16 v16bf;
typedef __attribute__((ext_vector_type(8)))  __bf16 v8bf;
typedef __attribute__((ext_vector_type(4)))  __bf16 v4bf;
typedef __attribute__((ext_vector_type(8)))  float  v8f;
typedef __attribute__((ext_vector_type(4)))  float  f32x4;

// --- kernel 1: csq[k] = ||c_k||^2 (fp32 exact); chi/clo = bf16 split ------
__global__ void prep_kernel(const float* __restrict__ c,
                            float* __restrict__ csq,
                            __bf16* __restrict__ chi,
                            __bf16* __restrict__ clo) {
  const int k = blockIdx.x;
  const int l = threadIdx.x;             // blockDim = 32 (one wave)
  const float* row  = c   + (size_t)k * DDIM;
  __bf16*      hrow = chi + (size_t)k * DDIM;
  __bf16*      lrow = clo + (size_t)k * DDIM;

  float s = 0.0f;
  #pragma unroll
  for (int g = 0; g < 4; ++g) {
    const int o = l * 16 + g * 4;        // lane owns 16 contiguous elems
    const f32x4 v = *(const f32x4*)(row + o);
    v4bf h, lo;
    #pragma unroll
    for (int e = 0; e < 4; ++e) {
      s += v[e] * v[e];
      const __bf16 hh = (__bf16)v[e];    // RNE
      h[e]  = hh;
      lo[e] = (__bf16)(v[e] - (float)hh);
    }
    *(v4bf*)(hrow + o) = h;
    *(v4bf*)(lrow + o) = lo;
  }
  s += __shfl_xor(s, 16);
  s += __shfl_xor(s, 8);
  s += __shfl_xor(s, 4);
  s += __shfl_xor(s, 2);
  s += __shfl_xor(s, 1);
  if (l == 0) csq[k] = s;
}

// --- kernel 2: fused GEMM (bf16 hi/lo split WMMA) + row softmax -----------
__global__ __launch_bounds__(512)
void proto_kernel(const float* __restrict__ x,
                  const float* __restrict__ csq,
                  const __bf16* __restrict__ chi,
                  const __bf16* __restrict__ clo,
                  float* __restrict__ out) {
  __shared__ float red[MROWS][NWAVES];
  __shared__ float rowstat[MROWS];

  const int tid     = threadIdx.x;
  const int w       = tid >> 5;          // wave id 0..15 (wave32)
  const int lane    = tid & 31;
  const int l15     = lane & 15;
  const int halfsel = lane >> 4;         // 0 / 1
  const int kb      = halfsel << 3;      // K sub-offset 0 or 8 (16-bit frag layout)
  const int rowbase = blockIdx.x * MROWS;

  v8f acc[2][4];                          // [m-tile][n-tile]
  #pragma unroll
  for (int mt = 0; mt < 2; ++mt)
    #pragma unroll
    for (int j = 0; j < 4; ++j)
      acc[mt][j] = (v8f){0.f, 0.f, 0.f, 0.f, 0.f, 0.f, 0.f, 0.f};

  const float* xrow0 = x + (size_t)(rowbase + l15) * DDIM;
  const float* xrow1 = xrow0 + (size_t)16 * DDIM;
  // per-lane base of this lane's B column (wave owns cols [w*64, w*64+64))
  const size_t colrow = (size_t)((w << 6) + l15) * DDIM;

  for (int step = 0; step < NSTEPS; ++step) {
    const int k0 = step * KC;

    // ---- A fragments for both M-tiles (16x32 bf16 layout):
    //      elems 0..7 = K kb..kb+7, elems 8..15 = K kb+16..kb+23 ----
    v16bf ahi[2], alo[2];
    #pragma unroll
    for (int mt = 0; mt < 2; ++mt) {
      const float* xr = mt ? xrow1 : xrow0;
      const f32x4 a0 = *(const f32x4*)(xr + k0 + kb);
      const f32x4 a1 = *(const f32x4*)(xr + k0 + kb + 4);
      const f32x4 a2 = *(const f32x4*)(xr + k0 + kb + 16);
      const f32x4 a3 = *(const f32x4*)(xr + k0 + kb + 20);
      float af[16];
      #pragma unroll
      for (int e = 0; e < 4; ++e) {
        af[e] = a0[e]; af[4 + e] = a1[e]; af[8 + e] = a2[e]; af[12 + e] = a3[e];
      }
      #pragma unroll
      for (int e = 0; e < 16; ++e) {
        const __bf16 h = (__bf16)af[e];
        ahi[mt][e] = h;
        alo[mt][e] = (__bf16)(af[e] - (float)h);
      }
    }

    // ---- 4 N-tiles: B frags straight from global bf16 planes (L2-hot);
    //      each feeds 6 WMMAs (2 M-tiles x hi*hi + hi*lo + lo*hi) ----
    #pragma unroll
    for (int j = 0; j < 4; ++j) {
      const size_t bo = colrow + (size_t)(j << 4) * DDIM + k0 + kb;
      const v8bf h0 = *(const v8bf*)(chi + bo);
      const v8bf h1 = *(const v8bf*)(chi + bo + 16);
      const v8bf l0 = *(const v8bf*)(clo + bo);
      const v8bf l1 = *(const v8bf*)(clo + bo + 16);
      const v16bf bhi = __builtin_shufflevector(h0, h1, 0, 1, 2, 3, 4, 5, 6, 7,
                                                8, 9, 10, 11, 12, 13, 14, 15);
      const v16bf blo = __builtin_shufflevector(l0, l1, 0, 1, 2, 3, 4, 5, 6, 7,
                                                8, 9, 10, 11, 12, 13, 14, 15);
      #pragma unroll
      for (int mt = 0; mt < 2; ++mt) {
        acc[mt][j] = __builtin_amdgcn_wmma_f32_16x16x32_bf16(
            false, ahi[mt], false, bhi, (short)0, acc[mt][j], false, false);
        acc[mt][j] = __builtin_amdgcn_wmma_f32_16x16x32_bf16(
            false, ahi[mt], false, blo, (short)0, acc[mt][j], false, false);
        acc[mt][j] = __builtin_amdgcn_wmma_f32_16x16x32_bf16(
            false, alo[mt], false, bhi, (short)0, acc[mt][j], false, false);
      }
    }
  }

  // ---- logits = 2*cross - csq[col] ----
  float cs[4];
  #pragma unroll
  for (int j = 0; j < 4; ++j)
    cs[j] = csq[(w << 6) + (j << 4) + l15];
  #pragma unroll
  for (int mt = 0; mt < 2; ++mt)
    #pragma unroll
    for (int j = 0; j < 4; ++j)
      #pragma unroll
      for (int i = 0; i < 8; ++i)
        acc[mt][j][i] = 2.0f * acc[mt][j][i] - cs[j];

  // ---- row max: C layout: VGPR i holds row (mt*16 + i + 8*halfsel)
  //      across a 16-lane half; reduce over j then shfl within the half ----
  #pragma unroll
  for (int mt = 0; mt < 2; ++mt) {
    #pragma unroll
    for (int i = 0; i < 8; ++i) {
      float m = acc[mt][0][i];
      #pragma unroll
      for (int j = 1; j < 4; ++j) m = fmaxf(m, acc[mt][j][i]);
      m = fmaxf(m, __shfl_xor(m, 1));
      m = fmaxf(m, __shfl_xor(m, 2));
      m = fmaxf(m, __shfl_xor(m, 4));
      m = fmaxf(m, __shfl_xor(m, 8));
      if (l15 == 0) red[(mt << 4) + (halfsel << 3) + i][w] = m;
    }
  }
  __syncthreads();
  if (tid < MROWS) {
    float m = red[tid][0];
    #pragma unroll
    for (int ww = 1; ww < NWAVES; ++ww) m = fmaxf(m, red[tid][ww]);
    rowstat[tid] = m;
  }
  __syncthreads();
  float rm[2][8];
  #pragma unroll
  for (int mt = 0; mt < 2; ++mt)
    #pragma unroll
    for (int i = 0; i < 8; ++i)
      rm[mt][i] = rowstat[(mt << 4) + (halfsel << 3) + i];
  __syncthreads();   // all reads of red/rowstat done before reuse

  // ---- exp + row sum ----
  #pragma unroll
  for (int mt = 0; mt < 2; ++mt) {
    #pragma unroll
    for (int i = 0; i < 8; ++i) {
      float s = 0.0f;
      #pragma unroll
      for (int j = 0; j < 4; ++j) {
        const float e = __expf(acc[mt][j][i] - rm[mt][i]);
        acc[mt][j][i] = e;
        s += e;
      }
      s += __shfl_xor(s, 1);
      s += __shfl_xor(s, 2);
      s += __shfl_xor(s, 4);
      s += __shfl_xor(s, 8);
      if (l15 == 0) red[(mt << 4) + (halfsel << 3) + i][w] = s;
    }
  }
  __syncthreads();
  if (tid < MROWS) {
    float s = 0.0f;
    #pragma unroll
    for (int ww = 0; ww < NWAVES; ++ww) s += red[tid][ww];
    rowstat[tid] = 1.0f / s;
  }
  __syncthreads();

  // ---- normalize + non-temporal store (output never re-read; keep L2 for B) ----
  #pragma unroll
  for (int mt = 0; mt < 2; ++mt) {
    #pragma unroll
    for (int i = 0; i < 8; ++i) {
      const int r = (mt << 4) + (halfsel << 3) + i;
      const float rinv = rowstat[r];
      float* orow = out + (size_t)(rowbase + r) * NCL;
      #pragma unroll
      for (int j = 0; j < 4; ++j)
        __builtin_nontemporal_store(acc[mt][j][i] * rinv,
                                    orow + (w << 6) + (j << 4) + l15);
    }
  }
}

// ---------------------------------------------------------------------------
extern "C" void kernel_launch(void* const* d_in, const int* in_sizes, int n_in,
                              void* d_out, int out_size, void* d_ws, size_t ws_size,
                              hipStream_t stream) {
  (void)in_sizes; (void)n_in; (void)out_size; (void)ws_size;
  const float* x        = (const float*)d_in[0];
  const float* clusters = (const float*)d_in[1];
  float*       out      = (float*)d_out;

  // d_ws layout: [csq: 1024 f32][chi: 1024*512 bf16][clo: 1024*512 bf16] ~2.1MB
  float*  csq = (float*)d_ws;
  __bf16* chi = (__bf16*)((char*)d_ws + 4096);
  __bf16* clo = chi + (size_t)NCL * DDIM;

  prep_kernel<<<NCL, 32, 0, stream>>>(clusters, csq, chi, clo);
  proto_kernel<<<BDIM / MROWS, 512, 0, stream>>>(x, csq, chi, clo, out);
}
